// MultiHeadAttention_60979945668694
// MI455X (gfx1250) — matrix operand order repaired and checked by reading, not compiled, not run on hardware
//
#include <hip/hip_runtime.h>
#include <hip/hip_bf16.h>

// ---------------------------------------------------------------------------
// MHA forward for MI455X (gfx1250), bf16 WMMA pipeline.
//   x:[2,2048,1024] f32, Wq/Wk/Wv/Wo:[1024,1024] f32 (out,in)  ->  out f32
// Pipeline: cvt -> 3x GEMM (Q,K,V bf16) -> flash-attn -> GEMM (Wo, f32 out)
// Data movement: TDM tensor_load_to_lds (TENSORcnt) with HW LDS padding,
// falling back to async global->LDS (ASYNCcnt), falling back to sync staging.
// ---------------------------------------------------------------------------

typedef __attribute__((ext_vector_type(16))) __bf16 bf16x16;
typedef __attribute__((ext_vector_type(8)))  float  f32x8;
typedef __attribute__((ext_vector_type(4)))  int    i32x4;

union FragAB { bf16x16 v; uint4 q[2]; };

#define D_MODEL 1024
#define SEQ_L   2048
#define NHEADS  16
#define DHEAD   64
#define MROWS   4096            // B * L

// ---- CDNA5 async global->LDS copy -----------------------------------------
#if defined(__has_builtin)
#if __has_builtin(__builtin_amdgcn_global_load_async_to_lds_b128) && \
    __has_builtin(__builtin_amdgcn_s_wait_asynccnt)
#define HAS_ASYNC_LDS 1
#endif
#if __has_builtin(__builtin_amdgcn_tensor_load_to_lds) && \
    __has_builtin(__builtin_amdgcn_s_wait_tensorcnt)
#define HAS_TDM 1
#endif
#endif

#if defined(HAS_TDM) && defined(__has_include)
#if __has_include(<hip/amd_detail/amd_gfx1250_TDM.h>)
#define TDM_6ARG 1   // amdgpu-toolchain (therock headers): 6-arg builtin
#endif
#endif

#ifdef HAS_ASYNC_LDS
typedef __attribute__((address_space(1))) i32x4 g_i32x4;
typedef __attribute__((address_space(3))) i32x4 l_i32x4;

__device__ __forceinline__ void async_cp_b128(const __bf16* g, __bf16* l) {
    __builtin_amdgcn_global_load_async_to_lds_b128(
        (g_i32x4*)g, (l_i32x4*)l, 0, 0);
}
__device__ __forceinline__ void wait_async0() {
    __builtin_amdgcn_s_wait_asynccnt(0);
}
#endif

#ifdef HAS_TDM
typedef __attribute__((ext_vector_type(4))) unsigned int u32x4;
typedef __attribute__((ext_vector_type(8))) int i32x8v;
typedef __attribute__((ext_vector_type(4))) int i32x4v;

__device__ __forceinline__ unsigned lds_addr_of(const void* p) {
    return (unsigned)(unsigned long long)
        (__attribute__((address_space(3))) const char*)p;
}

// 2D bf16 tile load: tile (tileW x tileH) from row-major tensor with row
// stride strideElems; HW inserts padAmount DWORDs into LDS every
// 2^(padInterval+1) DWORDs stored (turns packed rows into padded pitch).
__device__ __forceinline__ void tdm_load_2d_bf16(
    const __bf16* gaddr, unsigned ldsByte,
    unsigned tileW, unsigned tileH, unsigned strideElems,
    unsigned padIntervalCode, unsigned padAmountCode)
{
    unsigned long long ga = (unsigned long long)(size_t)gaddr;
    u32x4 g0;
    g0[0] = 1u;                                   // count=1, user descriptor
    g0[1] = ldsByte;                              // lds_addr
    g0[2] = (unsigned)ga;                         // global_addr[31:0]
    g0[3] = (unsigned)((ga >> 32) & 0x1FFFFFFu)   // global_addr[56:32]
          | (2u << 30);                           // type = 2 ("image")
    i32x8v g1;
    g1[0] = (int)((1u << 16)                      // data_size = 1 (2 bytes)
                | (1u << 20)                      // pad_enable
                | (padIntervalCode << 22)
                | (padAmountCode << 25));
    g1[1] = (int)((tileW & 0xFFFFu) << 16);       // tensor_dim0[15:0] @63:48
    g1[2] = (int)(((tileW >> 16) & 0xFFFFu)       // tensor_dim0[31:16]
                | ((tileH & 0xFFFFu) << 16));     // tensor_dim1[15:0]
    g1[3] = (int)(((tileH >> 16) & 0xFFFFu)       // tensor_dim1[31:16]
                | ((tileW & 0xFFFFu) << 16));     // tile_dim0
    g1[4] = (int)(tileH & 0xFFFFu);               // tile_dim1 ; tile_dim2=0
    g1[5] = (int)strideElems;                     // tensor_dim0_stride[31:0]
    g1[6] = 0;                                    // stride hi / dim1_stride lo
    g1[7] = 0;
    i32x4v z4 = {0, 0, 0, 0};
#ifdef TDM_6ARG
    i32x8v z8 = {0, 0, 0, 0, 0, 0, 0, 0};
    __builtin_amdgcn_tensor_load_to_lds(g0, g1, z4, z4, z8, 0);
#else
    __builtin_amdgcn_tensor_load_to_lds(g0, g1, z4, z4, 0);
#endif
}
__device__ __forceinline__ void wait_tensor0() {
    __builtin_amdgcn_s_wait_tensorcnt(0);
}
#endif

// ------------------------------ convert ------------------------------------
__global__ __launch_bounds__(256)
void cvt_f32_bf16(const float* __restrict__ in, __bf16* __restrict__ out, int n) {
    int i = (blockIdx.x * 256 + threadIdx.x) * 4;
    if (i + 3 < n) {
        float4 v = *reinterpret_cast<const float4*>(in + i);
        struct alignas(8) B4 { __bf16 e[4]; } b;
        b.e[0] = (__bf16)v.x; b.e[1] = (__bf16)v.y;
        b.e[2] = (__bf16)v.z; b.e[3] = (__bf16)v.w;
        *reinterpret_cast<B4*>(out + i) = b;
    }
}

// ------------------------------ GEMM: C = A * Bw^T -------------------------
// A:[M,K] bf16 row-major, Bw:[N,K] bf16 row-major (torch Linear weight).
// Block tile 128x128, BK=32, 256 threads (8 waves), wave tile 32x64.
// Double-buffered LDS staged by the TDM (one wave issues; TENSORcnt).
#define GPITCH 40   // bf16 elems per LDS row (80 B = 32 elems + 8 pad)

template<bool OUT_F32>
__global__ __launch_bounds__(256)
void gemm_bf16(const __bf16* __restrict__ A, const __bf16* __restrict__ Bw,
               float* __restrict__ Cf, __bf16* __restrict__ Cb,
               int M, int N, int K)
{
    __shared__ __bf16 As[2][128 * GPITCH];
    __shared__ __bf16 Bs[2][128 * GPITCH];

    const int tid  = threadIdx.x;
    const int lane = tid & 31;
    const int wave = tid >> 5;
    const int wm   = wave & 3;          // 4 waves along M (32 rows each)
    const int wn   = wave >> 2;         // 2 waves along N (64 cols each)

    const int rowBlk = blockIdx.y * 128;
    const int colBlk = blockIdx.x * 128;

    const int ldr = tid >> 1;           // 0..127 (tile row)
    const int ldc = (tid & 1) * 16;     // 0 or 16 (tile col)

    const int laneN   = lane & 15;
    const int halfOff = (lane >> 4) * 8;   // C-layout row offset
    const int laneK8  = (lane >> 4) * 8;   // A-frag K base
    const int laneK16 = (lane >> 4) * 16;  // unused; B-frags also use laneK8

    f32x8 acc[2][4];
#pragma unroll
    for (int mi = 0; mi < 2; ++mi)
#pragma unroll
        for (int ni = 0; ni < 4; ++ni)
#pragma unroll
            for (int r = 0; r < 8; ++r) acc[mi][ni][r] = 0.0f;

    const __bf16* aG = A  + (size_t)(rowBlk + ldr) * K + ldc;
    const __bf16* bG = Bw + (size_t)(colBlk + ldr) * K + ldc;
    (void)aG; (void)bG;

#if defined(HAS_TDM)
    // -------- TDM double-buffered pipeline (wave 0 drives the DMA) --------
    const __bf16* aT = A  + (size_t)rowBlk * K;   // tile origin (uniform)
    const __bf16* bT = Bw + (size_t)colBlk * K;
    if (wave == 0) {
        tdm_load_2d_bf16(aT, lds_addr_of(&As[0][0]), 32, 128, K, 3, 3);
        tdm_load_2d_bf16(bT, lds_addr_of(&Bs[0][0]), 32, 128, K, 3, 3);
        wait_tensor0();
    }
    __syncthreads();

    int buf = 0;
    for (int k0 = 0; k0 < K; k0 += 32) {
        if (k0 + 32 < K && wave == 0) {
            tdm_load_2d_bf16(aT + k0 + 32, lds_addr_of(&As[buf ^ 1][0]),
                             32, 128, K, 3, 3);
            tdm_load_2d_bf16(bT + k0 + 32, lds_addr_of(&Bs[buf ^ 1][0]),
                             32, 128, K, 3, 3);
        }

        FragAB af[2], bfr[4];
#pragma unroll
        for (int mi = 0; mi < 2; ++mi) {
            const __bf16* p = &As[buf][(wm * 32 + mi * 16 + laneN) * GPITCH + laneK8];
            af[mi].q[0] = *reinterpret_cast<const uint4*>(p);
            af[mi].q[1] = *reinterpret_cast<const uint4*>(p + 16);
        }
#pragma unroll
        for (int ni = 0; ni < 4; ++ni) {
            const __bf16* p = &Bs[buf][(wn * 64 + ni * 16 + laneN) * GPITCH + laneK8];
            const uint4* pq = reinterpret_cast<const uint4*>(p);
            bfr[ni].q[0] = pq[0];
            bfr[ni].q[1] = pq[2];
        }
#pragma unroll
        for (int mi = 0; mi < 2; ++mi)
#pragma unroll
            for (int ni = 0; ni < 4; ++ni)
                acc[mi][ni] = __builtin_amdgcn_wmma_f32_16x16x32_bf16(
                    false, af[mi].v, false, bfr[ni].v,
                    (short)0, acc[mi][ni], false, false);

        if (wave == 0) wait_tensor0();
        __syncthreads();
        buf ^= 1;
    }
#elif defined(HAS_ASYNC_LDS)
    // -------- async double-buffered pipeline --------
    {
        __bf16* ad = &As[0][ldr * GPITCH + ldc];
        __bf16* bd = &Bs[0][ldr * GPITCH + ldc];
        async_cp_b128(aG,     ad);
        async_cp_b128(aG + 8, ad + 8);
        async_cp_b128(bG,     bd);
        async_cp_b128(bG + 8, bd + 8);
    }
    wait_async0();
    __syncthreads();

    int buf = 0;
    for (int k0 = 0; k0 < K; k0 += 32) {
        if (k0 + 32 < K) {
            __bf16* ad = &As[buf ^ 1][ldr * GPITCH + ldc];
            __bf16* bd = &Bs[buf ^ 1][ldr * GPITCH + ldc];
            async_cp_b128(aG + k0 + 32, ad);
            async_cp_b128(aG + k0 + 40, ad + 8);
            async_cp_b128(bG + k0 + 32, bd);
            async_cp_b128(bG + k0 + 40, bd + 8);
        }

        FragAB af[2], bfr[4];
#pragma unroll
        for (int mi = 0; mi < 2; ++mi) {
            const __bf16* p = &As[buf][(wm * 32 + mi * 16 + laneN) * GPITCH + laneK8];
            af[mi].q[0] = *reinterpret_cast<const uint4*>(p);
            af[mi].q[1] = *reinterpret_cast<const uint4*>(p + 16);
        }
#pragma unroll
        for (int ni = 0; ni < 4; ++ni) {
            const __bf16* p = &Bs[buf][(wn * 64 + ni * 16 + laneN) * GPITCH + laneK8];
            const uint4* pq = reinterpret_cast<const uint4*>(p);
            bfr[ni].q[0] = pq[0];
            bfr[ni].q[1] = pq[2];
        }
#pragma unroll
        for (int mi = 0; mi < 2; ++mi)
#pragma unroll
            for (int ni = 0; ni < 4; ++ni)
                acc[mi][ni] = __builtin_amdgcn_wmma_f32_16x16x32_bf16(
                    false, af[mi].v, false, bfr[ni].v,
                    (short)0, acc[mi][ni], false, false);

        wait_async0();
        __syncthreads();
        buf ^= 1;
    }
#else
    // -------- fallback: synchronous staging --------
    for (int k0 = 0; k0 < K; k0 += 32) {
        const uint4* ag = reinterpret_cast<const uint4*>(aG + k0);
        uint4 a0 = ag[0], a1 = ag[1];
        const uint4* bg = reinterpret_cast<const uint4*>(bG + k0);
        uint4 b0 = bg[0], b1 = bg[1];

        __syncthreads();
        uint4* asd = reinterpret_cast<uint4*>(&As[0][ldr * GPITCH + ldc]);
        asd[0] = a0; asd[1] = a1;
        uint4* bsd = reinterpret_cast<uint4*>(&Bs[0][ldr * GPITCH + ldc]);
        bsd[0] = b0; bsd[1] = b1;
        __syncthreads();

        FragAB af[2], bfr[4];
#pragma unroll
        for (int mi = 0; mi < 2; ++mi) {
            const __bf16* p = &As[0][(wm * 32 + mi * 16 + laneN) * GPITCH + laneK8];
            af[mi].q[0] = *reinterpret_cast<const uint4*>(p);
            af[mi].q[1] = *reinterpret_cast<const uint4*>(p + 16);
        }
#pragma unroll
        for (int ni = 0; ni < 4; ++ni) {
            const __bf16* p = &Bs[0][(wn * 64 + ni * 16 + laneN) * GPITCH + laneK8];
            const uint4* pq = reinterpret_cast<const uint4*>(p);
            bfr[ni].q[0] = pq[0];
            bfr[ni].q[1] = pq[2];
        }
#pragma unroll
        for (int mi = 0; mi < 2; ++mi)
#pragma unroll
            for (int ni = 0; ni < 4; ++ni)
                acc[mi][ni] = __builtin_amdgcn_wmma_f32_16x16x32_bf16(
                    false, af[mi].v, false, bfr[ni].v,
                    (short)0, acc[mi][ni], false, false);
    }
#endif

    const int rBase = rowBlk + wm * 32;
    const int cBase = colBlk + wn * 64;
#pragma unroll
    for (int mi = 0; mi < 2; ++mi)
#pragma unroll
        for (int ni = 0; ni < 4; ++ni)
#pragma unroll
            for (int r = 0; r < 8; ++r) {
                int row = rBase + mi * 16 + r + halfOff;
                int col = cBase + ni * 16 + laneN;
                float v = acc[mi][ni][r];
                if (OUT_F32) Cf[(size_t)row * N + col] = v;
                else         Cb[(size_t)row * N + col] = (__bf16)v;
            }
}

// ------------------------------ Flash attention ----------------------------
// Q,K,V,O are [4096,1024] bf16; head h occupies cols [h*64, h*64+64).
// Grid: (L/64, B*H). 128 threads = 4 waves; wave owns 16 query rows.
#define APITCH 80   // bf16 elems per LDS row (160 B = 64 elems + 16 pad)

__global__ __launch_bounds__(128)
void attn_kernel(const __bf16* __restrict__ Q, const __bf16* __restrict__ Km,
                 const __bf16* __restrict__ Vm, __bf16* __restrict__ O)
{
    __shared__ __bf16 Ks[64 * APITCH];
    __shared__ __bf16 Vt[64 * APITCH];   // transposed: Vt[d][kv]
    __shared__ __bf16 Ps[64 * APITCH];   // per-wave private 16-row strips

    const int tid     = threadIdx.x;
    const int lane    = tid & 31;
    const int wave    = tid >> 5;        // 0..3
    const int laneN   = lane & 15;
    const int halfOff = (lane >> 4) * 8;
    const int laneK8  = (lane >> 4) * 8;
    const int laneK16 = (lane >> 4) * 16;

    const int qTile = blockIdx.x;        // 0..31
    const int bh    = blockIdx.y;        // 0..31
    const int b     = bh >> 4, h = bh & 15;
    const int qBase = qTile * 64;
    const size_t rowStride = D_MODEL;
    const size_t headOff   = (size_t)h * DHEAD;
    const size_t batchRow  = (size_t)b * SEQ_L;

    // Q strip for this wave: 16 rows x 64, two A-fragments (K = 0..31, 32..63)
    FragAB aQ[2];
    {
        const __bf16* qrow =
            Q + (batchRow + qBase + wave * 16 + laneN) * rowStride + headOff;
#pragma unroll
        for (int kc = 0; kc < 2; ++kc) {
            const __bf16* p = qrow + kc * 32 + laneK8;
            aQ[kc].q[0] = *reinterpret_cast<const uint4*>(p);
            aQ[kc].q[1] = *reinterpret_cast<const uint4*>(p + 16);
        }
    }

    f32x8 oAcc[4];
#pragma unroll
    for (int ni = 0; ni < 4; ++ni)
#pragma unroll
        for (int r = 0; r < 8; ++r) oAcc[ni][r] = 0.0f;

    float mrow[8], lrow[8];
#pragma unroll
    for (int r = 0; r < 8; ++r) { mrow[r] = -1e30f; lrow[r] = 0.0f; }

    const float scale = 0.125f;          // 1/sqrt(64)

    // staging coords (whole block cooperates)
    const int sr = tid >> 1;             // 0..63 kv row
    const int sc = (tid & 1) * 32;       // 0 or 32 (32 cols per thread)

    for (int kt = 0; kt <= qTile; ++kt) {
        const int kvBase = kt * 64;
        const __bf16* kg =
            Km + (batchRow + kvBase + sr) * rowStride + headOff + sc;
        const __bf16* vg =
            Vm + (batchRow + kvBase + sr) * rowStride + headOff + sc;

        // prefetch next tile while this one is consumed
        if (kt < qTile) {
            __builtin_prefetch(kg + 64 * rowStride, 0, 0);
            __builtin_prefetch(vg + 64 * rowStride, 0, 0);
        }

        // ---- stage K tile (row-major) and V tile (transposed) into LDS ----
        {
            // V comes through registers (needs element-wise transpose)
            const uint4* vq = reinterpret_cast<const uint4*>(vg);
            union { uint4 q[4]; __bf16 e[32]; } vv;
            vv.q[0] = vq[0]; vv.q[1] = vq[1]; vv.q[2] = vq[2]; vv.q[3] = vq[3];

#if defined(HAS_TDM)
            __syncthreads();   // previous iteration done reading LDS
            if (wave == 0) {
                tdm_load_2d_bf16(Km + (batchRow + kvBase) * rowStride + headOff,
                                 lds_addr_of(&Ks[0]), 64, 64, D_MODEL, 4, 7);
            }
#pragma unroll
            for (int j = 0; j < 32; ++j)
                Vt[(sc + j) * APITCH + sr] = vv.e[j];
            if (wave == 0) wait_tensor0();
            __syncthreads();
#elif defined(HAS_ASYNC_LDS)
            __syncthreads();   // previous iteration done reading LDS
            __bf16* kd = &Ks[sr * APITCH + sc];
            async_cp_b128(kg,      kd);
            async_cp_b128(kg + 8,  kd + 8);
            async_cp_b128(kg + 16, kd + 16);
            async_cp_b128(kg + 24, kd + 24);
#pragma unroll
            for (int j = 0; j < 32; ++j)
                Vt[(sc + j) * APITCH + sr] = vv.e[j];
            wait_async0();
            __syncthreads();
#else
            const uint4* kq = reinterpret_cast<const uint4*>(kg);
            uint4 k0 = kq[0], k1 = kq[1], k2 = kq[2], k3 = kq[3];
            __syncthreads();
            uint4* kd = reinterpret_cast<uint4*>(&Ks[sr * APITCH + sc]);
            kd[0] = k0; kd[1] = k1; kd[2] = k2; kd[3] = k3;
#pragma unroll
            for (int j = 0; j < 32; ++j)
                Vt[(sc + j) * APITCH + sr] = vv.e[j];
            __syncthreads();
#endif
        }

        // ---- S = (Q K^T) for this wave's 16 rows: 16 x 64 in 4 C-tiles ----
        f32x8 s[4];
#pragma unroll
        for (int ni = 0; ni < 4; ++ni)
#pragma unroll
            for (int r = 0; r < 8; ++r) s[ni][r] = 0.0f;
#pragma unroll
        for (int kc = 0; kc < 2; ++kc)
#pragma unroll
            for (int ni = 0; ni < 4; ++ni) {
                FragAB bk;
                const __bf16* p =
                    &Ks[(ni * 16 + laneN) * APITCH + kc * 32 + laneK8];
                const uint4* pq = reinterpret_cast<const uint4*>(p);
                bk.q[0] = pq[0];
                bk.q[1] = pq[2];
                s[ni] = __builtin_amdgcn_wmma_f32_16x16x32_bf16(
                    false, aQ[kc].v, false, bk.v, (short)0, s[ni], false, false);
            }

        // ---- scale + causal mask ----
        const bool diag = (kt == qTile);
#pragma unroll
        for (int ni = 0; ni < 4; ++ni) {
            int kv = kvBase + ni * 16 + laneN;
#pragma unroll
            for (int r = 0; r < 8; ++r) {
                float v = s[ni][r] * scale;
                if (diag) {
                    int qg = qBase + wave * 16 + r + halfOff;
                    if (kv > qg) v = -1e30f;
                }
                s[ni][r] = v;
            }
        }

        // ---- online softmax (row stats live per-lane for this half's rows) --
        float mnew[8], fsc[8];
#pragma unroll
        for (int r = 0; r < 8; ++r) {
            float mx = s[0][r];
#pragma unroll
            for (int ni = 1; ni < 4; ++ni) mx = fmaxf(mx, s[ni][r]);
            mx = fmaxf(mx, __shfl_xor(mx, 1, 32));
            mx = fmaxf(mx, __shfl_xor(mx, 2, 32));
            mx = fmaxf(mx, __shfl_xor(mx, 4, 32));
            mx = fmaxf(mx, __shfl_xor(mx, 8, 32));
            mnew[r] = fmaxf(mrow[r], mx);
            fsc[r]  = __expf(mrow[r] - mnew[r]);
            mrow[r] = mnew[r];
        }
#pragma unroll
        for (int ni = 0; ni < 4; ++ni)
#pragma unroll
            for (int r = 0; r < 8; ++r)
                s[ni][r] = __expf(s[ni][r] - mnew[r]);
#pragma unroll
        for (int r = 0; r < 8; ++r) {
            float sum = s[0][r] + s[1][r] + s[2][r] + s[3][r];
            sum += __shfl_xor(sum, 1, 32);
            sum += __shfl_xor(sum, 2, 32);
            sum += __shfl_xor(sum, 4, 32);
            sum += __shfl_xor(sum, 8, 32);
            lrow[r] = lrow[r] * fsc[r] + sum;
        }
#pragma unroll
        for (int ni = 0; ni < 4; ++ni)
#pragma unroll
            for (int r = 0; r < 8; ++r) oAcc[ni][r] *= fsc[r];

        // ---- C-layout -> A-layout for P via private LDS strip (intra-wave) --
        {
            __bf16* prow = &Ps[(size_t)wave * 16 * APITCH];
#pragma unroll
            for (int ni = 0; ni < 4; ++ni)
#pragma unroll
                for (int r = 0; r < 8; ++r)
                    prow[(r + halfOff) * APITCH + ni * 16 + laneN] =
                        (__bf16)s[ni][r];
        }
        FragAB aP[2];
#pragma unroll
        for (int kc = 0; kc < 2; ++kc) {
            const __bf16* p =
                &Ps[(wave * 16 + laneN) * APITCH + kc * 32 + laneK8];
            aP[kc].q[0] = *reinterpret_cast<const uint4*>(p);
            aP[kc].q[1] = *reinterpret_cast<const uint4*>(p + 16);
        }

        // ---- O += P * V (B-fragments from transposed Vt rows) ----
#pragma unroll
        for (int kc = 0; kc < 2; ++kc)
#pragma unroll
            for (int ni = 0; ni < 4; ++ni) {
                FragAB bv;
                const __bf16* p =
                    &Vt[(ni * 16 + laneN) * APITCH + kc * 32 + laneK8];
                const uint4* pq = reinterpret_cast<const uint4*>(p);
                bv.q[0] = pq[0];
                bv.q[1] = pq[2];
                oAcc[ni] = __builtin_amdgcn_wmma_f32_16x16x32_bf16(
                    false, aP[kc].v, false, bv.v, (short)0, oAcc[ni],
                    false, false);
            }
    }

    // ---- epilogue: normalize, store bf16 O ----
    {
        const int qRow = qBase + wave * 16;
#pragma unroll
        for (int ni = 0; ni < 4; ++ni)
#pragma unroll
            for (int r = 0; r < 8; ++r) {
                float v = oAcc[ni][r] / lrow[r];
                int qg = qRow + r + halfOff;
                O[(batchRow + qg) * rowStride + headOff + ni * 16 + laneN] =
                    (__bf16)v;
            }
    }
}

// ------------------------------ launcher -----------------------------------
extern "C" void kernel_launch(void* const* d_in, const int* in_sizes, int n_in,
                              void* d_out, int out_size, void* d_ws, size_t ws_size,
                              hipStream_t stream) {
    const float* x  = (const float*)d_in[0];
    const float* Wq = (const float*)d_in[1];
    const float* Wk = (const float*)d_in[2];
    const float* Wv = (const float*)d_in[3];
    const float* Wo = (const float*)d_in[4];
    float* out = (float*)d_out;

    char* ws = (char*)d_ws;
    __bf16* xb  = (__bf16*)(ws);                       // 8 MiB
    __bf16* wqb = (__bf16*)(ws + ( 8u << 20));         // 2 MiB each
    __bf16* wkb = (__bf16*)(ws + (10u << 20));
    __bf16* wvb = (__bf16*)(ws + (12u << 20));
    __bf16* wob = (__bf16*)(ws + (14u << 20));
    __bf16* Qb  = (__bf16*)(ws + (16u << 20));         // 8 MiB each
    __bf16* Kb  = (__bf16*)(ws + (24u << 20));
    __bf16* Vb  = (__bf16*)(ws + (32u << 20));
    __bf16* Ob  = (__bf16*)(ws + (40u << 20));

    const int nX = MROWS * D_MODEL;        // 4 M elements
    const int nW = D_MODEL * D_MODEL;      // 1 M elements
    cvt_f32_bf16<<<(nX / 4 + 255) / 256, 256, 0, stream>>>(x,  xb,  nX);
    cvt_f32_bf16<<<(nW / 4 + 255) / 256, 256, 0, stream>>>(Wq, wqb, nW);
    cvt_f32_bf16<<<(nW / 4 + 255) / 256, 256, 0, stream>>>(Wk, wkb, nW);
    cvt_f32_bf16<<<(nW / 4 + 255) / 256, 256, 0, stream>>>(Wv, wvb, nW);
    cvt_f32_bf16<<<(nW / 4 + 255) / 256, 256, 0, stream>>>(Wo, wob, nW);

    dim3 ggrid(D_MODEL / 128, MROWS / 128);            // (8, 32)
    gemm_bf16<false><<<ggrid, 256, 0, stream>>>(xb, wqb, nullptr, Qb,
                                                MROWS, D_MODEL, D_MODEL);
    gemm_bf16<false><<<ggrid, 256, 0, stream>>>(xb, wkb, nullptr, Kb,
                                                MROWS, D_MODEL, D_MODEL);
    gemm_bf16<false><<<ggrid, 256, 0, stream>>>(xb, wvb, nullptr, Vb,
                                                MROWS, D_MODEL, D_MODEL);

    dim3 agrid(SEQ_L / 64, 2 * NHEADS);                // (32, 32)
    attn_kernel<<<agrid, 128, 0, stream>>>(Qb, Kb, Vb, Ob);

    gemm_bf16<true><<<ggrid, 256, 0, stream>>>(Ob, wob, out, nullptr,
                                               MROWS, D_MODEL, D_MODEL);
}
